// DSN_TEMPORAL_266287972624
// MI455X (gfx1250) — compile-verified, matrix operand order checked by reading, not run on hardware
//
#include <hip/hip_runtime.h>
#include <hip/hip_bf16.h>
#include <math.h>

// ---------------------------------------------------------------------------
// Problem sizes (fixed by the reference):
//   support: [64, 8, 576]  -> 512  rows of 576
//   query:   [2048, 8, 576]-> 16384 rows of 576
//   frame_dists: [16384, 512] ; dists: [2048, 64, 8, 8] ; out: [2048, 64]
// ---------------------------------------------------------------------------
#define D_DIM      576
#define NQ_ROWS    16384   // 2048*8
#define NS_ROWS    512     // 64*8
#define N_PAIRS    131072  // 2048*64
#define LBDA_INV   2.0f    // 1/lambda, lambda = 0.5
#define LBDA       0.5f

typedef __bf16 v16bf __attribute__((ext_vector_type(16)));
typedef __bf16 v8bf  __attribute__((ext_vector_type(8)));
typedef float  v8f   __attribute__((ext_vector_type(8)));

// ---------------------------------------------------------------------------
// Kernel 1: per-row L2 norm + f32 -> bf16 conversion (RNE). One wave per row.
// ---------------------------------------------------------------------------
__global__ void norm_convert_kernel(const float* __restrict__ in,
                                    unsigned short* __restrict__ out_bf,
                                    float* __restrict__ out_norm,
                                    int nrows)
{
    int wave = (int)((blockIdx.x * blockDim.x + threadIdx.x) >> 5);
    int lane = (int)(threadIdx.x & 31);
    if (wave >= nrows) return;

    const float* row = in + (size_t)wave * D_DIM;
    unsigned short* orow = out_bf + (size_t)wave * D_DIM;

    float ss = 0.0f;
    #pragma unroll
    for (int c = 0; c < D_DIM / 32; ++c) {           // 576/32 = 18 iterations
        int idx = lane + c * 32;
        float v = row[idx];
        ss = fmaf(v, v, ss);
        unsigned int b = __float_as_uint(v);
        unsigned int r = (b + 0x7FFFu + ((b >> 16) & 1u)) >> 16;  // RNE to bf16
        orow[idx] = (unsigned short)r;
    }
    // wave32 butterfly reduction
    #pragma unroll
    for (int off = 16; off > 0; off >>= 1)
        ss += __shfl_xor(ss, off, 32);
    if (lane == 0) out_norm[wave] = sqrtf(ss);
}

// ---------------------------------------------------------------------------
// Kernel 2: bf16 WMMA GEMM  C[i,j] = q[i,:] . s[j,:]   (16384 x 512, K = 576)
// One wave per 16x16 tile, 18 steps of v_wmma_f32_16x16x32_bf16.
// Epilogue: dist = 1 - C/(|q||s| + 0.01), scattered into [2048,64,8,8].
// ---------------------------------------------------------------------------
__global__ void __launch_bounds__(256)
wmma_gemm_dist_kernel(const unsigned short* __restrict__ qbf,
                      const unsigned short* __restrict__ sbf,
                      const float* __restrict__ qn,
                      const float* __restrict__ sn,
                      float* __restrict__ dists)
{
    const int NT = 32;                         // 512/16 N-tiles
    int wave = (int)(threadIdx.x >> 5);
    int lane = (int)(threadIdx.x & 31);
    int tile = (int)blockIdx.x * 8 + wave;     // 4096 blocks * 8 waves = 32768 tiles
    int mt = tile / NT;
    int nt = tile % NT;

    int mrow = lane & 15;                      // M (for A) / N (for B) within tile
    int kb   = (lane >> 4) << 3;               // 0 or 8: k-base of this half-wave

    const __bf16* arow = reinterpret_cast<const __bf16*>(
        qbf + (size_t)(mt * 16 + mrow) * D_DIM + kb);
    const __bf16* brow = reinterpret_cast<const __bf16*>(
        sbf + (size_t)(nt * 16 + mrow) * D_DIM + kb);

    v8f acc = {};
    #pragma unroll
    for (int k0 = 0; k0 < D_DIM; k0 += 32) {   // 18 WMMA steps
        // Prefetch next iteration's A/B stream (global_prefetch_b8).
        __builtin_prefetch(arow + k0 + 32, 0, 0);
        __builtin_prefetch(brow + k0 + 32, 0, 0);

        v8bf alo = *(const v8bf*)(arow + k0);        // K = k0+kb   .. +7
        v8bf ahi = *(const v8bf*)(arow + k0 + 16);   // K = k0+16+kb.. +7
        v8bf blo = *(const v8bf*)(brow + k0);
        v8bf bhi = *(const v8bf*)(brow + k0 + 16);

        v16bf A = __builtin_shufflevector(alo, ahi, 0, 1, 2, 3, 4, 5, 6, 7,
                                          8, 9, 10, 11, 12, 13, 14, 15);
        v16bf B = __builtin_shufflevector(blo, bhi, 0, 1, 2, 3, 4, 5, 6, 7,
                                          8, 9, 10, 11, 12, 13, 14, 15);

        acc = __builtin_amdgcn_wmma_f32_16x16x32_bf16(
            /*neg_a=*/false, A, /*neg_b=*/false, B,
            /*c_mod=*/(short)0, acc, /*reuse_a=*/false, /*reuse_b=*/false);
    }

    // C/D layout: VGPR r, lanes 0-15 -> M=r, N=lane; lanes 16-31 -> M=r+8, N=lane-16
    int N = lane & 15;
    int halfsel = lane >> 4;
    int j = nt * 16 + N;
    float snj = sn[j];
    #pragma unroll
    for (int r = 0; r < 8; ++r) {
        int i = mt * 16 + halfsel * 8 + r;
        float val = 1.0f - acc[r] / (qn[i] * snj + 0.01f);
        // dists[i>>3][j>>3][i&7][j&7], shape [2048][64][8][8]
        size_t idx = (((size_t)(i >> 3) * 64 + (size_t)(j >> 3)) << 6)
                   + (size_t)((i & 7) * 8 + (j & 7));
        dists[idx] = val;
    }
}

// ---------------------------------------------------------------------------
// Kernel 3: OTAM soft-DTW DP, one (query, support) pair per thread.
// Runs the 8x10 recurrence forward and on the transposed block; fully
// unrolled so the 64-float block stays in VGPRs.
// ---------------------------------------------------------------------------
__device__ __forceinline__ float softmin_step(float d, float diag, float left,
                                              float up, bool has_up)
{
    float mn = fminf(diag, left);
    if (has_up) mn = fminf(mn, up);
    float s = __expf((mn - diag) * LBDA_INV) + __expf((mn - left) * LBDA_INV);
    if (has_up) s += __expf((mn - up) * LBDA_INV);
    return d + mn - LBDA * __logf(s);
}

__global__ void __launch_bounds__(256)
otam_dp_kernel(const float* __restrict__ dists, float* __restrict__ out)
{
    int p = (int)(blockIdx.x * blockDim.x + threadIdx.x);
    if (p >= N_PAIRS) return;

    float db[64];
    const float4* src = (const float4*)(dists + (size_t)p * 64);
    #pragma unroll
    for (int i = 0; i < 16; ++i) {
        float4 v = src[i];
        db[4 * i + 0] = v.x; db[4 * i + 1] = v.y;
        db[4 * i + 2] = v.z; db[4 * i + 3] = v.w;
    }

    float total = 0.0f;
    #pragma unroll
    for (int ori = 0; ori < 2; ++ori) {
        // D(l,s): forward = db[l*8+s]; transposed = db[s*8+l]
        float cum[10];
        cum[0] = 0.0f;
        #pragma unroll
        for (int m = 1; m < 10; ++m) {
            float d0 = (m <= 8) ? (ori ? db[(m - 1) * 8 + 0] : db[0 * 8 + (m - 1)])
                                : 0.0f;
            cum[m] = cum[m - 1] + d0;
        }
        #pragma unroll
        for (int l = 1; l < 8; ++l) {
            float nrow[10];
            nrow[0] = 0.0f;
            float left = 0.0f;
            #pragma unroll
            for (int m = 1; m < 10; ++m) {
                float dval = (m <= 8) ? (ori ? db[(m - 1) * 8 + l] : db[l * 8 + (m - 1)])
                                      : 0.0f;
                bool has_up = (m == 1) || (m == 9);
                float nv = softmin_step(dval, cum[m - 1], left, cum[m], has_up);
                nrow[m] = nv;
                left = nv;
            }
            #pragma unroll
            for (int m = 0; m < 10; ++m) cum[m] = nrow[m];
        }
        total += cum[9];
    }
    out[p] = -total;
}

// ---------------------------------------------------------------------------
// Launcher. Workspace layout (bytes):
//   [0)           qbf   16384*576*2 = 18874368
//   [18874368)    sbf     512*576*2 =   589824
//   [19464192)    qn    16384*4     =    65536
//   [19529728)    sn      512*4     =     2048
//   [19531776)    dists 16384*512*4 = 33554432   (total ~50.6 MB)
// ---------------------------------------------------------------------------
extern "C" void kernel_launch(void* const* d_in, const int* in_sizes, int n_in,
                              void* d_out, int out_size, void* d_ws, size_t ws_size,
                              hipStream_t stream)
{
    (void)in_sizes; (void)n_in; (void)out_size; (void)ws_size;

    const float* support = (const float*)d_in[0];  // [64, 8, 576]
    const float* query   = (const float*)d_in[1];  // [2048, 8, 576]
    float* out = (float*)d_out;                    // [2048, 64]

    char* ws = (char*)d_ws;
    unsigned short* qbf = (unsigned short*)(ws);
    unsigned short* sbf = (unsigned short*)(ws + 18874368);
    float* qn    = (float*)(ws + 19464192);
    float* sn    = (float*)(ws + 19529728);
    float* dists = (float*)(ws + 19531776);

    // 1) norms + bf16 conversion: one wave per row, 8 waves/block
    norm_convert_kernel<<<NQ_ROWS / 8, 256, 0, stream>>>(query, qbf, qn, NQ_ROWS);
    norm_convert_kernel<<<NS_ROWS / 8, 256, 0, stream>>>(support, sbf, sn, NS_ROWS);

    // 2) WMMA GEMM + cosine-distance epilogue: 32768 tiles, 8 waves/block
    wmma_gemm_dist_kernel<<<4096, 256, 0, stream>>>(qbf, sbf, qn, sn, dists);

    // 3) OTAM DP: one pair per thread
    otam_dp_kernel<<<N_PAIRS / 256, 256, 0, stream>>>(dists, out);
}
